// MPNNModel_970662609384
// MI455X (gfx1250) — matrix operand order repaired and checked by reading, not compiled, not run on hardware
//
#include <hip/hip_runtime.h>
#include <hip/hip_bf16.h>

#define N_NODES 100000
#define N_EDGES 1000000
#define IN_DIM  49
#define EMB     64
#define N_LAYERS 4
#define BN_EPS  1e-5f
#define EDGE_TILES (N_EDGES / 16)   // 62500
#define NODE_TILES (N_NODES / 16)   // 6250

typedef __attribute__((ext_vector_type(16))) __bf16 v16bf;
typedef __attribute__((ext_vector_type(8)))  __bf16 v8bf;
typedef __attribute__((ext_vector_type(8)))  float  v8f;

__device__ __forceinline__ __bf16 f2bf(float f) {
    unsigned u = __float_as_uint(f);
    u += 0x7fffu + ((u >> 16) & 1u);           // RNE to bf16
    unsigned short h = (unsigned short)(u >> 16);
    __bf16 r;
    __builtin_memcpy(&r, &h, 2);
    return r;
}

__device__ __forceinline__ float bf2f(__bf16 b) {
    unsigned short h;
    __builtin_memcpy(&h, &b, 2);
    return __uint_as_float(((unsigned)h) << 16);
}

__device__ __forceinline__ void atomAddF(float* p, float v) {
    __hip_atomic_fetch_add(p, v, __ATOMIC_RELAXED, __HIP_MEMORY_SCOPE_AGENT);
}

__device__ __forceinline__ v8f wmma_bf16(v16bf a, v16bf b, v8f c) {
    return __builtin_amdgcn_wmma_f32_16x16x32_bf16(false, a, false, b, (short)0, c, false, false);
}

// B fragment (32x16 tile of W[K][64] row-major, fp32 -> bf16).
// col=N=lane%16, hi=lane/16; elem e<8: K (kbase + 8*hi + e); e>=8: K (kbase + 16 + 8*hi + e-8).
__device__ __forceinline__ v16bf load_bfrag(const float* __restrict__ W, int kbase, int ncolbase, int lane) {
    int l16 = lane & 15, hi = (lane >> 4) & 1;
    int col = ncolbase + l16;
    v16bf b;
#pragma unroll
    for (int e = 0; e < 16; ++e)
        b[e] = f2bf(W[(kbase + ((e < 8) ? (hi * 8 + e) : (16 + hi * 8 + (e - 8)))) * EMB + col]);
    return b;
}

// A fragment from bf16 row (p points at row start + K-chunk base col).
__device__ __forceinline__ v16bf load_afrag_bf16(const __bf16* __restrict__ p, int hi) {
    v8bf lo  = *reinterpret_cast<const v8bf*>(p + hi * 8);
    v8bf hi8 = *reinterpret_cast<const v8bf*>(p + 16 + hi * 8);
    return __builtin_shufflevector(lo, hi8, 0,1,2,3,4,5,6,7,8,9,10,11,12,13,14,15);
}

// A fragment from bf16 row with per-element f32 affine + ReLU (BN apply).
__device__ __forceinline__ v16bf afrag_bnrelu_bf16(const __bf16* __restrict__ p, int hi,
                                                   const float aR[16], const float bR[16]) {
    v8bf lo  = *reinterpret_cast<const v8bf*>(p + hi * 8);
    v8bf hi8 = *reinterpret_cast<const v8bf*>(p + 16 + hi * 8);
    v16bf a;
#pragma unroll
    for (int e = 0; e < 8; ++e) {
        float m = fmaxf(aR[e] * bf2f(lo[e]) + bR[e], 0.f);
        a[e] = f2bf(m);
    }
#pragma unroll
    for (int e = 0; e < 8; ++e) {
        float m = fmaxf(aR[8 + e] * bf2f(hi8[e]) + bR[8 + e], 0.f);
        a[8 + e] = f2bf(m);
    }
    return a;
}

// A fragment from fp32 row (aggr), plain convert.
__device__ __forceinline__ v16bf afrag_from_f32(const float* __restrict__ p, int hi) {
    const float4* q0 = reinterpret_cast<const float4*>(p + hi * 8);
    const float4* q1 = reinterpret_cast<const float4*>(p + 16 + hi * 8);
    float4 u0 = q0[0], u1 = q0[1], u2 = q1[0], u3 = q1[1];
    float vals[16] = {u0.x,u0.y,u0.z,u0.w, u1.x,u1.y,u1.z,u1.w,
                      u2.x,u2.y,u2.z,u2.w, u3.x,u3.y,u3.z,u3.w};
    v16bf a;
#pragma unroll
    for (int e = 0; e < 16; ++e) a[e] = f2bf(vals[e]);
    return a;
}

// ---------------- input embedding: h = x @ W_in + b_in ----------------
__global__ void k_embed(const float* __restrict__ x, const float* __restrict__ Win,
                        const float* __restrict__ bin, float* __restrict__ h,
                        __bf16* __restrict__ hbf) {
    __shared__ float xr[IN_DIM];
    int node = blockIdx.x, t = threadIdx.x;        // 64 threads
    if (t < IN_DIM) xr[t] = x[node * IN_DIM + t];
    __syncthreads();
    float acc = bin[t];
#pragma unroll 7
    for (int k = 0; k < IN_DIM; ++k) acc = fmaf(xr[k], Win[k * EMB + t], acc);
    h[node * EMB + t]   = acc;
    hbf[node * EMB + t] = f2bf(acc);
}

// -------- edge GEMM1: z1 = cat[h[dst],h[src],ea] @ W1 + b1, + BN stats --------
__global__ void __launch_bounds__(128)
k_edge_gemm1(const __bf16* __restrict__ hbf, const int* __restrict__ srcI,
             const int* __restrict__ dstI, const float* __restrict__ ea,
             const float* __restrict__ W1 /*[129][64]*/, const float* __restrict__ b1,
             __bf16* __restrict__ z1, float* __restrict__ sSum, float* __restrict__ sSq) {
    int lane = threadIdx.x & 31, l16 = lane & 15, hi = (lane >> 4) & 1;
    int wave = blockIdx.x * (blockDim.x >> 5) + (threadIdx.x >> 5);
    int nwaves = gridDim.x * (blockDim.x >> 5);

    v16bf Bf[4][4];
#pragma unroll
    for (int c = 0; c < 4; ++c)
#pragma unroll
        for (int n = 0; n < 4; ++n) Bf[c][n] = load_bfrag(W1, 32 * c, 16 * n, lane);

    float wrow[4], bcol[4];
#pragma unroll
    for (int n = 0; n < 4; ++n) {
        wrow[n] = W1[128 * EMB + 16 * n + l16];   // edge_attr rank-1 row
        bcol[n] = b1[16 * n + l16];
    }
    float s[4] = {0,0,0,0}, q[4] = {0,0,0,0};

    for (int t = wave; t < EDGE_TILES; t += nwaves) {
        int e0 = t * 16;
        int d  = dstI[e0 + l16];
        int sr = srcI[e0 + l16];
        float eaL = ea[e0 + l16];
        v8f acc[4];
#pragma unroll
        for (int n = 0; n < 4; ++n)
#pragma unroll
            for (int r = 0; r < 8; ++r) acc[n][r] = 0.f;

#pragma unroll
        for (int c = 0; c < 4; ++c) {
            int row = (c < 2) ? d : sr;
            v16bf A = load_afrag_bf16(hbf + row * EMB + 32 * (c & 1), hi);
#pragma unroll
            for (int n = 0; n < 4; ++n) acc[n] = wmma_bf16(A, Bf[c][n], acc[n]);
        }
#pragma unroll
        for (int n = 0; n < 4; ++n)
#pragma unroll
            for (int r = 0; r < 8; ++r) {
                float eaM = __shfl(eaL, hi * 8 + r, 32);
                float v = acc[n][r] + eaM * wrow[n] + bcol[n];
                s[n] += v; q[n] += v * v;
                z1[(e0 + hi * 8 + r) * EMB + 16 * n + l16] = f2bf(v);
            }
    }
#pragma unroll
    for (int n = 0; n < 4; ++n) {
        float ss = s[n] + __shfl_xor(s[n], 16, 32);
        float qq = q[n] + __shfl_xor(q[n], 16, 32);
        if (lane < 16) {
            atomAddF(&sSum[16 * n + l16], ss);
            atomAddF(&sSq[16 * n + l16], qq);
        }
    }
}

// ------ generic GEMM2: zout = relu(a*zin+b) @ W2 + b2, + BN stats ------
__global__ void __launch_bounds__(128)
k_gemm2(const __bf16* __restrict__ zin, const float* __restrict__ aS, const float* __restrict__ bS,
        const float* __restrict__ W2 /*[64][64]*/, const float* __restrict__ b2,
        __bf16* __restrict__ zout, float* __restrict__ sSum, float* __restrict__ sSq, int ntiles) {
    int lane = threadIdx.x & 31, l16 = lane & 15, hi = (lane >> 4) & 1;
    int wave = blockIdx.x * (blockDim.x >> 5) + (threadIdx.x >> 5);
    int nwaves = gridDim.x * (blockDim.x >> 5);

    v16bf Bf[2][4];
#pragma unroll
    for (int c = 0; c < 2; ++c)
#pragma unroll
        for (int n = 0; n < 4; ++n) Bf[c][n] = load_bfrag(W2, 32 * c, 16 * n, lane);

    float aReg[2][16], bReg[2][16];
#pragma unroll
    for (int c = 0; c < 2; ++c)
#pragma unroll
        for (int e = 0; e < 16; ++e) {
            int col = 32 * c + ((e < 8) ? (hi * 8 + e) : (16 + hi * 8 + (e - 8)));
            aReg[c][e] = aS[col]; bReg[c][e] = bS[col];
        }
    float bcol[4];
#pragma unroll
    for (int n = 0; n < 4; ++n) bcol[n] = b2[16 * n + l16];

    float s[4] = {0,0,0,0}, q[4] = {0,0,0,0};
    for (int t = wave; t < ntiles; t += nwaves) {
        const __bf16* zr = zin + (t * 16 + l16) * EMB;
        v8f acc[4];
#pragma unroll
        for (int n = 0; n < 4; ++n)
#pragma unroll
            for (int r = 0; r < 8; ++r) acc[n][r] = 0.f;
#pragma unroll
        for (int c = 0; c < 2; ++c) {
            v16bf A = afrag_bnrelu_bf16(zr + 32 * c, hi, aReg[c], bReg[c]);
#pragma unroll
            for (int n = 0; n < 4; ++n) acc[n] = wmma_bf16(A, Bf[c][n], acc[n]);
        }
#pragma unroll
        for (int n = 0; n < 4; ++n)
#pragma unroll
            for (int r = 0; r < 8; ++r) {
                float v = acc[n][r] + bcol[n];
                s[n] += v; q[n] += v * v;
                zout[(t * 16 + hi * 8 + r) * EMB + 16 * n + l16] = f2bf(v);
            }
    }
#pragma unroll
    for (int n = 0; n < 4; ++n) {
        float ss = s[n] + __shfl_xor(s[n], 16, 32);
        float qq = q[n] + __shfl_xor(q[n], 16, 32);
        if (lane < 16) {
            atomAddF(&sSum[16 * n + l16], ss);
            atomAddF(&sSq[16 * n + l16], qq);
        }
    }
}

// ------ node GEMM1: zu1 = cat[h, aggr] @ uW1 + ub1, + BN stats ------
__global__ void __launch_bounds__(128)
k_node_gemm1(const __bf16* __restrict__ hbf, const float* __restrict__ aggr,
             const float* __restrict__ W1 /*[128][64]*/, const float* __restrict__ b1,
             __bf16* __restrict__ zout, float* __restrict__ sSum, float* __restrict__ sSq) {
    int lane = threadIdx.x & 31, l16 = lane & 15, hi = (lane >> 4) & 1;
    int wave = blockIdx.x * (blockDim.x >> 5) + (threadIdx.x >> 5);
    int nwaves = gridDim.x * (blockDim.x >> 5);

    v16bf Bf[4][4];
#pragma unroll
    for (int c = 0; c < 4; ++c)
#pragma unroll
        for (int n = 0; n < 4; ++n) Bf[c][n] = load_bfrag(W1, 32 * c, 16 * n, lane);

    float bcol[4];
#pragma unroll
    for (int n = 0; n < 4; ++n) bcol[n] = b1[16 * n + l16];

    float s[4] = {0,0,0,0}, q[4] = {0,0,0,0};
    for (int t = wave; t < NODE_TILES; t += nwaves) {
        int r0 = t * 16 + l16;
        v8f acc[4];
#pragma unroll
        for (int n = 0; n < 4; ++n)
#pragma unroll
            for (int r = 0; r < 8; ++r) acc[n][r] = 0.f;
#pragma unroll
        for (int c = 0; c < 4; ++c) {
            v16bf A = (c < 2) ? load_afrag_bf16(hbf + r0 * EMB + 32 * c, hi)
                              : afrag_from_f32(aggr + r0 * EMB + 32 * (c - 2), hi);
#pragma unroll
            for (int n = 0; n < 4; ++n) acc[n] = wmma_bf16(A, Bf[c][n], acc[n]);
        }
#pragma unroll
        for (int n = 0; n < 4; ++n)
#pragma unroll
            for (int r = 0; r < 8; ++r) {
                float v = acc[n][r] + bcol[n];
                s[n] += v; q[n] += v * v;
                zout[(t * 16 + hi * 8 + r) * EMB + 16 * n + l16] = f2bf(v);
            }
    }
#pragma unroll
    for (int n = 0; n < 4; ++n) {
        float ss = s[n] + __shfl_xor(s[n], 16, 32);
        float qq = q[n] + __shfl_xor(q[n], 16, 32);
        if (lane < 16) {
            atomAddF(&sSum[16 * n + l16], ss);
            atomAddF(&sSq[16 * n + l16], qq);
        }
    }
}

// ------ finalize BN: a = g*rsqrt(var+eps), b = be - mean*a ------
__global__ void k_finalize(const float* __restrict__ sSum, const float* __restrict__ sSq,
                           const float* __restrict__ g, const float* __restrict__ be,
                           float* __restrict__ aO, float* __restrict__ bO, float invCnt) {
    int i = threadIdx.x;
    float mean = sSum[i] * invCnt;
    float var  = fmaxf(sSq[i] * invCnt - mean * mean, 0.f);
    float a = g[i] * rsqrtf(var + BN_EPS);
    aO[i] = a;
    bO[i] = be[i] - mean * a;
}

// ------ scatter-add: aggr[dst] += relu(a*z2+b) ------
__global__ void k_scatter(const __bf16* __restrict__ z2, const float* __restrict__ aS,
                          const float* __restrict__ bS, const int* __restrict__ dstI,
                          float* __restrict__ aggr) {
    int stride = gridDim.x * blockDim.x;
    for (int i = blockIdx.x * blockDim.x + threadIdx.x; i < N_EDGES * EMB; i += stride) {
        int e = i >> 6, col = i & 63;
        float m = fmaxf(aS[col] * bf2f(z2[i]) + bS[col], 0.f);
        atomAddF(&aggr[dstI[e] * EMB + col], m);
    }
}

// ------ residual update: h += relu(a*zu2+b); refresh bf16 copy ------
__global__ void k_update_h(float* __restrict__ h, __bf16* __restrict__ hbf,
                           const __bf16* __restrict__ zu2, const float* __restrict__ aS,
                           const float* __restrict__ bS) {
    int stride = gridDim.x * blockDim.x;
    for (int i = blockIdx.x * blockDim.x + threadIdx.x; i < N_NODES * EMB; i += stride) {
        int col = i & 63;
        float hn = h[i] + fmaxf(aS[col] * bf2f(zu2[i]) + bS[col], 0.f);
        h[i] = hn;
        hbf[i] = f2bf(hn);
    }
}

// ------ column sums of h (for graph mean) ------
__global__ void k_colsum(const float* __restrict__ h, float* __restrict__ colsum) {
    __shared__ float red[256];
    int gid = blockIdx.x * blockDim.x + threadIdx.x;
    int stride = gridDim.x * blockDim.x;     // multiple of 64 -> col fixed per thread
    int col = gid & 63;
    float ls = 0.f;
    for (int i = gid; i < N_NODES * EMB; i += stride) ls += h[i];
    red[threadIdx.x] = ls;
    __syncthreads();
    if (threadIdx.x < 64) {
        float tot = red[threadIdx.x] + red[threadIdx.x + 64] +
                    red[threadIdx.x + 128] + red[threadIdx.x + 192];
        atomAddF(&colsum[col], tot);
    }
}

// ------ out = mean(h) @ W_pred + b_pred ------
__global__ void k_final(const float* __restrict__ colsum, const float* __restrict__ Wp,
                        const float* __restrict__ bp, float* __restrict__ out) {
    __shared__ float red[64];
    int t = threadIdx.x;
    red[t] = colsum[t] * (1.0f / N_NODES) * Wp[t];
    __syncthreads();
    for (int s = 32; s > 0; s >>= 1) {
        if (t < s) red[t] += red[t + s];
        __syncthreads();
    }
    if (t == 0) out[0] = red[0] + bp[0];
}

extern "C" void kernel_launch(void* const* d_in, const int* in_sizes, int n_in,
                              void* d_out, int out_size, void* d_ws, size_t ws_size,
                              hipStream_t stream) {
    const float* x       = (const float*)d_in[0];
    const int*   eidx    = (const int*)d_in[1];
    const float* ea      = (const float*)d_in[2];
    const float* W_in    = (const float*)d_in[3];
    const float* b_in    = (const float*)d_in[4];
    const float* msg_W1  = (const float*)d_in[5];
    const float* msg_b1  = (const float*)d_in[6];
    const float* msg_g1  = (const float*)d_in[7];
    const float* msg_be1 = (const float*)d_in[8];
    const float* msg_W2  = (const float*)d_in[9];
    const float* msg_b2  = (const float*)d_in[10];
    const float* msg_g2  = (const float*)d_in[11];
    const float* msg_be2 = (const float*)d_in[12];
    const float* upd_W1  = (const float*)d_in[13];
    const float* upd_b1  = (const float*)d_in[14];
    const float* upd_g1  = (const float*)d_in[15];
    const float* upd_be1 = (const float*)d_in[16];
    const float* upd_W2  = (const float*)d_in[17];
    const float* upd_b2  = (const float*)d_in[18];
    const float* upd_g2  = (const float*)d_in[19];
    const float* upd_be2 = (const float*)d_in[20];
    const float* W_pred  = (const float*)d_in[21];
    const float* b_pred  = (const float*)d_in[22];
    const int* srcI = eidx;             // edge_index[0]
    const int* dstI = eidx + N_EDGES;   // edge_index[1]

    // workspace layout
    float* stats  = (float*)d_ws;            // 1024 floats reserved
    float* sum1 = stats;       float* sq1 = stats + 64;
    float* aS1  = stats + 128; float* bS1 = stats + 192;
    float* sum2 = stats + 256; float* sq2 = stats + 320;
    float* aS2  = stats + 384; float* bS2 = stats + 448;
    float* colsum = stats + 512;
    char* big = (char*)d_ws + 4096;
    float*  h    = (float*)(big);                       //  25.6 MB
    __bf16* hbf  = (__bf16*)(big + 25600000);           //  12.8 MB
    float*  aggr = (float*)(big + 38400000);            //  25.6 MB
    __bf16* z1   = (__bf16*)(big + 64000000);           // 128 MB (bf16, aliased as zu1)
    __bf16* z2   = (__bf16*)(big + 192000000);          // 128 MB (bf16, aliased as zu2)

    k_embed<<<N_NODES, 64, 0, stream>>>(x, W_in, b_in, h, hbf);

    for (int l = 0; l < N_LAYERS; ++l) {
        const float* mW1 = msg_W1 + l * 129 * EMB;
        const float* mW2 = msg_W2 + l * EMB * EMB;
        const float* uW1 = upd_W1 + l * 128 * EMB;
        const float* uW2 = upd_W2 + l * EMB * EMB;

        // message MLP layer 1
        hipMemsetAsync(sum1, 0, 512, stream);
        k_edge_gemm1<<<1024, 128, 0, stream>>>(hbf, srcI, dstI, ea, mW1, msg_b1 + l * EMB,
                                               z1, sum1, sq1);
        k_finalize<<<1, 64, 0, stream>>>(sum1, sq1, msg_g1 + l * EMB, msg_be1 + l * EMB,
                                         aS1, bS1, 1.0f / N_EDGES);
        // message MLP layer 2
        hipMemsetAsync(sum2, 0, 512, stream);
        k_gemm2<<<1024, 128, 0, stream>>>(z1, aS1, bS1, mW2, msg_b2 + l * EMB,
                                          z2, sum2, sq2, EDGE_TILES);
        k_finalize<<<1, 64, 0, stream>>>(sum2, sq2, msg_g2 + l * EMB, msg_be2 + l * EMB,
                                         aS2, bS2, 1.0f / N_EDGES);
        // aggregate
        hipMemsetAsync(aggr, 0, (size_t)N_NODES * EMB * sizeof(float), stream);
        k_scatter<<<2048, 256, 0, stream>>>(z2, aS2, bS2, dstI, aggr);

        // update MLP layer 1 (zu1 aliases z1)
        hipMemsetAsync(sum1, 0, 512, stream);
        k_node_gemm1<<<256, 128, 0, stream>>>(hbf, aggr, uW1, upd_b1 + l * EMB,
                                              z1, sum1, sq1);
        k_finalize<<<1, 64, 0, stream>>>(sum1, sq1, upd_g1 + l * EMB, upd_be1 + l * EMB,
                                         aS1, bS1, 1.0f / N_NODES);
        // update MLP layer 2 (zu2 aliases z2)
        hipMemsetAsync(sum2, 0, 512, stream);
        k_gemm2<<<256, 128, 0, stream>>>(z1, aS1, bS1, uW2, upd_b2 + l * EMB,
                                         z2, sum2, sq2, NODE_TILES);
        k_finalize<<<1, 64, 0, stream>>>(sum2, sq2, upd_g2 + l * EMB, upd_be2 + l * EMB,
                                         aS2, bS2, 1.0f / N_NODES);
        // residual
        k_update_h<<<1024, 256, 0, stream>>>(h, hbf, z2, aS2, bS2);
    }

    hipMemsetAsync(colsum, 0, 256, stream);
    k_colsum<<<512, 256, 0, stream>>>(h, colsum);
    k_final<<<1, 64, 0, stream>>>(colsum, W_pred, b_pred, (float*)d_out);
}
